// GRU_Decoder_22522808500368
// MI455X (gfx1250) — matrix operand order repaired and checked by reading, not compiled, not run on hardware
//
#include <hip/hip_runtime.h>
#include <hip/hip_bf16.h>
#include <math.h>

// ---------------------------------------------------------------------------
// GRU decoder, CDNA5 (gfx1250): bf16 WMMA 16x16x32, f32 accumulate.
// Sizes fixed by the reference.
// ---------------------------------------------------------------------------
#define BB   64      // batch
#define TT   128     // time steps
#define EE   1024    // hidden / embed
#define KK   1024    // reduction dim (D_IN == V_IN == E)

typedef __attribute__((ext_vector_type(16))) __bf16 v16bf;
typedef __attribute__((ext_vector_type(8)))  float  v8f;

__device__ __forceinline__ float sigmoid_f(float x) {
  return 1.0f / (1.0f + __expf(-x));
}

#define WMMA_BF16(a, b, c) \
  __builtin_amdgcn_wmma_f32_16x16x32_bf16(false, (a), false, (b), (short)0, (c), false, false)

// ---------------------------------------------------------------------------
// Pack a row-major f32 weight matrix [K x N] into WMMA-B bf16 fragments (this code's contiguous-k order).
// Fragment layout (per 32-K-tile x 16-N-tile):
//   lane = n_local | ((kk>>4)<<4)   (lanes 0-15: K=0..15, lanes 16-31: K=16..31)
//   elem j = kk & 15
// Stored so each lane's 16 bf16 (32B) are contiguous -> 2x global_load_b128.
// ---------------------------------------------------------------------------
__global__ void pack_b_kernel(const float* __restrict__ W, __bf16* __restrict__ P,
                              int K, int N) {
  int tid = blockIdx.x * blockDim.x + threadIdx.x;   // K * (N/16) threads
  int ntiles = N >> 4;
  int k = tid / ntiles;
  int ntile = tid - k * ntiles;
  if (k >= K) return;
  int ktiles  = K >> 5;
  int ktile   = k >> 5;
  int kk      = k & 31;
  int lane_hi = (kk >> 4) << 4;
  int j       = kk & 15;
  const float* src = W + (size_t)k * N + ((size_t)ntile << 4);
  __bf16* dst = P + ((size_t)(ntile * ktiles + ktile) << 9);   // *512 elems
#pragma unroll
  for (int n = 0; n < 16; ++n) {
    dst[(size_t)(n | lane_hi) * 16 + j] = (__bf16)src[n];
  }
}

// Load a 16x32 A fragment from a row-major f32 matrix (row = lane&15).
// A layout: lanes 0-15 & 16-31 both hold rows M=0..15; lane[4] picks K-half.
// elems 0..15 -> K = k0 + khalf + 0..15 (khalf = 0 or 16)
__device__ __forceinline__ v16bf load_a_frag(const float* __restrict__ Arow,
                                             int k0, int lane) {
  int khalf = (lane >> 4) << 4;
  const float4* p = (const float4*)(Arow + k0 + khalf);
  float4 f0 = p[0], f1 = p[1], f2 = p[2], f3 = p[3];
  v16bf a;
  a[0]  = (__bf16)f0.x; a[1]  = (__bf16)f0.y; a[2]  = (__bf16)f0.z; a[3]  = (__bf16)f0.w;
  a[4]  = (__bf16)f1.x; a[5]  = (__bf16)f1.y; a[6]  = (__bf16)f1.z; a[7]  = (__bf16)f1.w;
  a[8]  = (__bf16)f2.x; a[9]  = (__bf16)f2.y; a[10] = (__bf16)f2.z; a[11] = (__bf16)f2.w;
  a[12] = (__bf16)f3.x; a[13] = (__bf16)f3.y; a[14] = (__bf16)f3.z; a[15] = (__bf16)f3.w;
  return a;
}

// ---------------------------------------------------------------------------
// Generic GEMM: Out[M,N] = A[M,K](f32) @ Bp(packed bf16) (+ bias[N]).
// Each wave computes a 32x32 output block (2 M-tiles x 2 N-tiles, 4 WMMAs
// per k-tile, 2x reuse of both A and B fragments). 8 waves / 256-thr block.
// Requires M % 32 == 0, N % 32 == 0 (holds: M in {8192, 64}, N = 1024).
// ---------------------------------------------------------------------------
__global__ void gemm_bias_kernel(const float* __restrict__ A,
                                 const __bf16* __restrict__ Bp,
                                 const float* __restrict__ bias,
                                 float* __restrict__ Out,
                                 int M, int N, int K) {
  int warp = blockIdx.x * (blockDim.x >> 5) + (threadIdx.x >> 5);
  int lane = threadIdx.x & 31;
  int nblk = N >> 5, mblk = M >> 5, ktiles = K >> 5;
  if (warp >= mblk * nblk) return;
  int mb = warp / nblk;
  int nb = warp - mb * nblk;

  const float*  Arow0 = A + (size_t)((mb << 5) + (lane & 15)) * K;
  const float*  Arow1 = Arow0 + (size_t)16 * K;
  const __bf16* B0 = Bp + ((size_t)((2 * nb)     * ktiles) << 9) + (size_t)lane * 16;
  const __bf16* B1 = Bp + ((size_t)((2 * nb + 1) * ktiles) << 9) + (size_t)lane * 16;

  v8f acc00 = {}, acc01 = {}, acc10 = {}, acc11 = {};
  for (int kt = 0; kt < ktiles; ++kt) {
    v16bf a0 = load_a_frag(Arow0, kt << 5, lane);
    v16bf a1 = load_a_frag(Arow1, kt << 5, lane);
    v16bf b0 = *(const v16bf*)(B0 + ((size_t)kt << 9));
    v16bf b1 = *(const v16bf*)(B1 + ((size_t)kt << 9));
    acc00 = WMMA_BF16(a0, b0, acc00);
    acc01 = WMMA_BF16(a0, b1, acc01);
    acc10 = WMMA_BF16(a1, b0, acc10);
    acc11 = WMMA_BF16(a1, b1, acc11);
  }
  // D layout: elem v -> (m = mtile*16 + v + 8*(lane>>4), n = ntile*16 + lane&15)
  int n0    = (nb << 5) + (lane & 15);
  int mbase = (mb << 5) + ((lane >> 4) << 3);
  float bv0 = bias ? bias[n0]      : 0.0f;
  float bv1 = bias ? bias[n0 + 16] : 0.0f;
#pragma unroll
  for (int v = 0; v < 8; ++v) {
    size_t r0 = (size_t)(mbase + v) * N;
    size_t r1 = (size_t)(mbase + 16 + v) * N;
    Out[r0 + n0]      = acc00[v] + bv0;
    Out[r0 + n0 + 16] = acc01[v] + bv1;
    Out[r1 + n0]      = acc10[v] + bv0;
    Out[r1 + n0 + 16] = acc11[v] + bv1;
  }
}

__global__ void zero_kernel(float* __restrict__ p, int n) {
  int i = blockIdx.x * blockDim.x + threadIdx.x;
  if (i < n) p[i] = 0.0f;
}

// ---------------------------------------------------------------------------
// Recurrence phase 1 (step t):
//   z  = sigmoid(xz[:,t,:] + h @ U_z + vz)
//   r  = sigmoid(xr[:,t,:] + h @ U_r + vr)
//   rh = r * h
// One wave -> one 16x16 tile of both gates (A fragment of h shared).
// Latency-bound: keep max wave count (256 waves) rather than blocking.
// ---------------------------------------------------------------------------
__global__ void gru_phase1_kernel(const float* __restrict__ h,
                                  const __bf16* __restrict__ Uzp,
                                  const __bf16* __restrict__ Urp,
                                  const float* __restrict__ xz,
                                  const float* __restrict__ xr,
                                  const float* __restrict__ vz,
                                  const float* __restrict__ vr,
                                  float* __restrict__ zbuf,
                                  float* __restrict__ rhbuf,
                                  int t) {
  int warp = blockIdx.x * (blockDim.x >> 5) + (threadIdx.x >> 5);
  int lane = threadIdx.x & 31;
  const int ntiles = EE >> 4, ktiles = EE >> 5;
  if (warp >= (BB >> 4) * ntiles) return;
  int mtile = warp / ntiles;
  int ntile = warp - mtile * ntiles;

  const float*  Arow = h + (size_t)((mtile << 4) + (lane & 15)) * EE;
  const __bf16* Bz   = Uzp + ((size_t)(ntile * ktiles) << 9) + (size_t)lane * 16;
  const __bf16* Br   = Urp + ((size_t)(ntile * ktiles) << 9) + (size_t)lane * 16;

  v8f accz = {}, accr = {};
  for (int kt = 0; kt < ktiles; ++kt) {
    v16bf a  = load_a_frag(Arow, kt << 5, lane);
    v16bf bz = *(const v16bf*)(Bz + ((size_t)kt << 9));
    v16bf br = *(const v16bf*)(Br + ((size_t)kt << 9));
    accz = WMMA_BF16(a, bz, accz);
    accr = WMMA_BF16(a, br, accr);
  }

  int n     = (ntile << 4) + (lane & 15);
  int mbase = (mtile << 4) + ((lane >> 4) << 3);
#pragma unroll
  for (int v = 0; v < 8; ++v) {
    int m = mbase + v;
    size_t pe = (size_t)m * EE + n;            // [B,E] element
    size_t px = ((size_t)m * TT + t) * EE + n; // [B,T,E] element
    float zv = sigmoid_f(accz[v] + xz[px] + vz[pe]);
    float rv = sigmoid_f(accr[v] + xr[px] + vr[pe]);
    zbuf[pe]  = zv;
    rhbuf[pe] = rv * h[pe];
  }
}

// ---------------------------------------------------------------------------
// Recurrence phase 2 (step t):
//   hh    = tanh(xh[:,t,:] + rh @ U + vh)
//   h_new = (1-z)*h + z*hh      (written to h and to out[:,t,:])
// ---------------------------------------------------------------------------
__global__ void gru_phase2_kernel(float* __restrict__ h,
                                  const __bf16* __restrict__ Up,
                                  const float* __restrict__ xh,
                                  const float* __restrict__ vh,
                                  const float* __restrict__ zbuf,
                                  const float* __restrict__ rhbuf,
                                  float* __restrict__ out,
                                  int t) {
  int warp = blockIdx.x * (blockDim.x >> 5) + (threadIdx.x >> 5);
  int lane = threadIdx.x & 31;
  const int ntiles = EE >> 4, ktiles = EE >> 5;
  if (warp >= (BB >> 4) * ntiles) return;
  int mtile = warp / ntiles;
  int ntile = warp - mtile * ntiles;

  const float*  Arow = rhbuf + (size_t)((mtile << 4) + (lane & 15)) * EE;
  const __bf16* Bu   = Up + ((size_t)(ntile * ktiles) << 9) + (size_t)lane * 16;

  v8f acc = {};
  for (int kt = 0; kt < ktiles; ++kt) {
    v16bf a = load_a_frag(Arow, kt << 5, lane);
    v16bf b = *(const v16bf*)(Bu + ((size_t)kt << 9));
    acc = WMMA_BF16(a, b, acc);
  }

  int n     = (ntile << 4) + (lane & 15);
  int mbase = (mtile << 4) + ((lane >> 4) << 3);
#pragma unroll
  for (int v = 0; v < 8; ++v) {
    int m = mbase + v;
    size_t pe = (size_t)m * EE + n;
    size_t px = ((size_t)m * TT + t) * EE + n;
    float hhv = tanhf(acc[v] + xh[px] + vh[pe]);
    float zv  = zbuf[pe];
    float hn  = (1.0f - zv) * h[pe] + zv * hhv;
    h[pe]   = hn;            // each element owned by exactly one lane
    out[px] = hn;            // out is [B,T,E]
  }
}

// ---------------------------------------------------------------------------
// Host-side orchestration (graph-capture safe: only kernel launches).
// ---------------------------------------------------------------------------
extern "C" void kernel_launch(void* const* d_in, const int* in_sizes, int n_in,
                              void* d_out, int out_size, void* d_ws, size_t ws_size,
                              hipStream_t stream) {
  (void)in_sizes; (void)n_in; (void)out_size; (void)ws_size;
  const float* video = (const float*)d_in[0];
  const float* seq   = (const float*)d_in[1];
  const float* W_z   = (const float*)d_in[2];
  const float* U_z   = (const float*)d_in[3];
  const float* C_z   = (const float*)d_in[4];
  const float* b_z   = (const float*)d_in[5];
  const float* W_r   = (const float*)d_in[6];
  const float* U_r   = (const float*)d_in[7];
  const float* C_r   = (const float*)d_in[8];
  const float* b_r   = (const float*)d_in[9];
  const float* W     = (const float*)d_in[10];
  const float* U     = (const float*)d_in[11];
  const float* C     = (const float*)d_in[12];
  const float* b     = (const float*)d_in[13];
  float* out = (float*)d_out;

  // Workspace carve-out
  char* ws = (char*)d_ws;
  size_t off = 0;
  auto carve = [&](size_t bytes) -> void* {
    void* p = ws + off;
    off += (bytes + 255) & ~(size_t)255;
    return p;
  };
  const size_t WBYTES = (size_t)KK * EE * sizeof(__bf16);   // 2 MB packed
  __bf16* Wz_p = (__bf16*)carve(WBYTES);
  __bf16* Wr_p = (__bf16*)carve(WBYTES);
  __bf16* W_p  = (__bf16*)carve(WBYTES);
  __bf16* Uz_p = (__bf16*)carve(WBYTES);
  __bf16* Ur_p = (__bf16*)carve(WBYTES);
  __bf16* U_p  = (__bf16*)carve(WBYTES);
  __bf16* Cz_p = (__bf16*)carve(WBYTES);
  __bf16* Cr_p = (__bf16*)carve(WBYTES);
  __bf16* C_p  = (__bf16*)carve(WBYTES);

  const size_t XBYTES = (size_t)BB * TT * EE * sizeof(float);  // 32 MB
  float* xz = (float*)carve(XBYTES);
  float* xr = (float*)carve(XBYTES);
  float* xh = (float*)carve(XBYTES);

  const size_t VBYTES = (size_t)BB * EE * sizeof(float);       // 256 KB
  float* vz    = (float*)carve(VBYTES);
  float* vr    = (float*)carve(VBYTES);
  float* vh    = (float*)carve(VBYTES);
  float* zbuf  = (float*)carve(VBYTES);
  float* rhbuf = (float*)carve(VBYTES);
  float* hbuf  = (float*)carve(VBYTES);

  // 1) Pack all 9 weight matrices into WMMA-B bf16 fragments.
  {
    int threads = KK * (EE >> 4);                 // 65536
    dim3 g(threads / 256), blk(256);
    pack_b_kernel<<<g, blk, 0, stream>>>(W_z, Wz_p, KK, EE);
    pack_b_kernel<<<g, blk, 0, stream>>>(W_r, Wr_p, KK, EE);
    pack_b_kernel<<<g, blk, 0, stream>>>(W,   W_p,  KK, EE);
    pack_b_kernel<<<g, blk, 0, stream>>>(U_z, Uz_p, EE, EE);
    pack_b_kernel<<<g, blk, 0, stream>>>(U_r, Ur_p, EE, EE);
    pack_b_kernel<<<g, blk, 0, stream>>>(U,   U_p,  EE, EE);
    pack_b_kernel<<<g, blk, 0, stream>>>(C_z, Cz_p, KK, EE);
    pack_b_kernel<<<g, blk, 0, stream>>>(C_r, Cr_p, KK, EE);
    pack_b_kernel<<<g, blk, 0, stream>>>(C,   C_p,  KK, EE);
  }

  // 2) Zero h0.
  zero_kernel<<<(BB * EE) / 256, 256, 0, stream>>>(hbuf, BB * EE);

  // 3) Time-invariant projections (32x32 per wave).
  {
    // seq @ W_*  : M = B*T = 8192, no bias
    int waves  = (BB * TT >> 5) * (EE >> 5);      // 8192
    dim3 g(waves / 8), blk(256);
    gemm_bias_kernel<<<g, blk, 0, stream>>>(seq, Wz_p, nullptr, xz, BB * TT, EE, KK);
    gemm_bias_kernel<<<g, blk, 0, stream>>>(seq, Wr_p, nullptr, xr, BB * TT, EE, KK);
    gemm_bias_kernel<<<g, blk, 0, stream>>>(seq, W_p,  nullptr, xh, BB * TT, EE, KK);
    // video @ C_* + b_* : M = 64
    int waves2 = (BB >> 5) * (EE >> 5);           // 64
    dim3 g2(waves2 / 8);
    gemm_bias_kernel<<<g2, blk, 0, stream>>>(video, Cz_p, b_z, vz, BB, EE, KK);
    gemm_bias_kernel<<<g2, blk, 0, stream>>>(video, Cr_p, b_r, vr, BB, EE, KK);
    gemm_bias_kernel<<<g2, blk, 0, stream>>>(video, C_p,  b,   vh, BB, EE, KK);
  }

  // 4) Sequential GRU recurrence: 2 dependent phases per step.
  {
    int waves = (BB >> 4) * (EE >> 4);            // 256 tiles
    dim3 g(waves / 8), blk(256);
    for (int t = 0; t < TT; ++t) {
      gru_phase1_kernel<<<g, blk, 0, stream>>>(hbuf, Uz_p, Ur_p, xz, xr,
                                               vz, vr, zbuf, rhbuf, t);
      gru_phase2_kernel<<<g, blk, 0, stream>>>(hbuf, U_p, xh, vh,
                                               zbuf, rhbuf, out, t);
    }
  }
}
